// L2Attention_61375082660372
// MI455X (gfx1250) — matrix operand order repaired and checked by reading, not compiled, not run on hardware
//
#include <hip/hip_runtime.h>
#include <hip/hip_bf16.h>

// ---------------------------------------------------------------------------
// L2-distance attention, MI455X (gfx1250, wave32, WMMA bf16 path)
//   B=2, T=2048, D=1024, H=16, hd=64, scale = hd^-0.5 = 0.125
// Pipeline: convert->bf16, QKV gemm (WMMA, V written pre-transposed),
//           ||q||^2, flash attention (WMMA), projection gemm (WMMA + bias)
// All global->LDS staging uses CDNA5 async-to-LDS, double-buffered so the
// next tile's DMA overlaps the current tile's WMMA work.
// ---------------------------------------------------------------------------

typedef __attribute__((ext_vector_type(16))) __bf16 v16bf;
typedef __attribute__((ext_vector_type(8)))  __bf16 v8bf;
typedef __attribute__((ext_vector_type(8)))  float  v8f;
typedef __attribute__((ext_vector_type(4)))  int    v4i;

union V16U { v16bf v; v8bf h[2]; };

__device__ __forceinline__ v8f wmma_bf16(v16bf a, v16bf b, v8f c) {
  // D = A(16x32) * B(32x16) + C(16x16 f32)
  return __builtin_amdgcn_wmma_f32_16x16x32_bf16(
      /*neg_a=*/false, a, /*neg_b=*/false, b,
      /*c_mod=*/(short)0, c, /*reuse_a=*/false, /*reuse_b=*/false);
}

// A-operand fragment: lane holds row (row0 + lane%16); K split pattern:
// lanes 0-15: K = k0+{0..7, 16..23}; lanes 16-31: K = k0+{8..15, 24..31}
__device__ __forceinline__ v16bf frag_a(const __bf16* p, int ld, int row0,
                                        int k0, int lane) {
  int r   = row0 + (lane & 15);
  int klo = k0 + ((lane & 16) ? 8 : 0);
  const __bf16* q = p + (size_t)r * ld + klo;
  V16U u;
  u.h[0] = *(const v8bf*)(q);
  u.h[1] = *(const v8bf*)(q + 16);
  return u.v;
}

// B-operand fragment from a [N,K] row-major tile (i.e. B^T storage):
// lane holds column n = row0 + lane%16; lanes 0-15: K = k0+{0..7, 16..23},
// lanes 16-31: K = k0+{8..15, 24..31} (same split as frag_a)
__device__ __forceinline__ v16bf frag_b(const __bf16* p, int ld, int row0,
                                        int k0, int lane) {
  int r  = row0 + (lane & 15);
  int kb = k0 + ((lane & 16) ? 8 : 0);
  const __bf16* q = p + (size_t)r * ld + kb;
  V16U u;
  u.h[0] = *(const v8bf*)(q);
  u.h[1] = *(const v8bf*)(q + 16);
  return u.v;
}

// ---------------------------------------------------------------------------
// Global -> LDS staging via CDNA5 async-to-LDS (ASYNCcnt path, no VGPR
// round-trip); falls back to load+store if builtin is absent.
#if __has_builtin(__builtin_amdgcn_global_load_async_to_lds_b128)
#define HAVE_ASYNC_LDS 1
#endif

__device__ __forceinline__ void stage_b128(const __bf16* g, __bf16* l) {
#ifdef HAVE_ASYNC_LDS
  __builtin_amdgcn_global_load_async_to_lds_b128(
      (__attribute__((address_space(1))) v4i*)
          (__attribute__((address_space(1))) void*)g,
      (__attribute__((address_space(3))) v4i*)
          (__attribute__((address_space(3))) void*)l,
      0, 0);
#else
  *(v8bf*)l = *(const v8bf*)g;
#endif
}

__device__ __forceinline__ void stage_wait() {
#ifdef HAVE_ASYNC_LDS
  asm volatile("s_wait_asynccnt 0" ::: "memory");
#endif
}

// ---------------------------------------------------------------------------
__global__ __launch_bounds__(256) void convert_bf16(const float* __restrict__ s,
                                                    __bf16* __restrict__ d, int n) {
  int i = blockIdx.x * 256 + threadIdx.x;
  if (i < n) d[i] = (__bf16)s[i];
}

// sq[b,h,t] = sum_d Q[b,h,t,d]^2  (one thread per (b,h,t))
__global__ __launch_bounds__(256) void sq_kernel(const __bf16* __restrict__ Q,
                                                 float* __restrict__ sq) {
  int i = blockIdx.x * 256 + threadIdx.x;          // (b*H+h)*T + t
  const v8bf* q = (const v8bf*)(Q + (size_t)i * 64);
  float s = 0.f;
  #pragma unroll
  for (int c = 0; c < 8; ++c) {
    v8bf v = q[c];
    #pragma unroll
    for (int e = 0; e < 8; ++e) { float f = (float)v[e]; s += f * f; }
  }
  sq[i] = s;
}

// ---------------------------------------------------------------------------
// C[M,N] = A[M,1024] @ W[N,1024]^T ; block tile 128x64, 128 threads = 4 waves,
// each wave: 32 rows x 64 cols (2 A-frags x 4 B-frags = 8 WMMA per K-step).
// Double-buffered async staging: DMA of K-step k+32 overlaps WMMA of step k.
// MODE 0: scatter qkv -> Q [B,H,T,hd] and V^T [B,H,hd,T] (bf16).
// MODE 1: +bias -> f32 out.
template <int MODE>
__global__ __launch_bounds__(128) void gemm128x64(
    const __bf16* __restrict__ A, const __bf16* __restrict__ W,
    __bf16* __restrict__ Qout, __bf16* __restrict__ Vout,
    const float* __restrict__ bias, float* __restrict__ Cout) {
  constexpr int K = 1024;
  __shared__ __bf16 As[2][128 * 32];
  __shared__ __bf16 Ws[2][64 * 32];
  const int tid = threadIdx.x;
  const int wave = tid >> 5, lane = tid & 31;
  const int m0 = blockIdx.y * 128, n0 = blockIdx.x * 64;

  v8f acc[2][4];
  #pragma unroll
  for (int i = 0; i < 2; ++i)
    #pragma unroll
    for (int j = 0; j < 4; ++j)
      #pragma unroll
      for (int r = 0; r < 8; ++r) acc[i][j][r] = 0.f;

  auto stage = [&](int k0, int b) {
    #pragma unroll
    for (int i = 0; i < 4; ++i) {                   // A: 128x32 bf16
      int idx = tid + i * 128;
      int r = idx >> 2, c = (idx & 3) * 8;
      stage_b128(&A[(size_t)(m0 + r) * K + k0 + c], &As[b][r * 32 + c]);
    }
    #pragma unroll
    for (int i = 0; i < 2; ++i) {                   // W: 64x32 bf16
      int idx = tid + i * 128;
      int r = idx >> 2, c = (idx & 3) * 8;
      stage_b128(&W[(size_t)(n0 + r) * K + k0 + c], &Ws[b][r * 32 + c]);
    }
  };
  auto compute = [&](int b) {
    v16bf a0 = frag_a(As[b], 32, wave * 32, 0, lane);
    v16bf a1 = frag_a(As[b], 32, wave * 32 + 16, 0, lane);
    #pragma unroll
    for (int j = 0; j < 4; ++j) {
      v16bf bb = frag_b(Ws[b], 32, j * 16, 0, lane);
      acc[0][j] = wmma_bf16(a0, bb, acc[0][j]);
      acc[1][j] = wmma_bf16(a1, bb, acc[1][j]);
    }
  };

  stage(0, 0);                               // prologue: fill buffer 0
  for (int k0 = 0; k0 < K; k0 += 64) {
    stage_wait();                            // buf0 data (issued last phase)
    __syncthreads();                         // all waves done reading buf1
    stage(k0 + 32, 1);                       // DMA next step into buf1
    compute(0);                              // WMMA on buf0 meanwhile
    stage_wait();
    __syncthreads();
    if (k0 + 64 < K) stage(k0 + 64, 0);
    compute(1);
  }

  const int hi = lane >> 4, lo = lane & 15;
  #pragma unroll
  for (int i = 0; i < 2; ++i)
    #pragma unroll
    for (int j = 0; j < 4; ++j)
      #pragma unroll
      for (int r = 0; r < 8; ++r) {
        int m = m0 + wave * 32 + i * 16 + r + 8 * hi;  // D: m = r + 8*(lane/16)
        int n = n0 + j * 16 + lo;                      //    n = lane%16
        float v = acc[i][j][r];
        if (MODE == 0) {
          // qkv column n decomposes as e = d*32 + k*16 + h (reference reshape)
          int b = m >> 11, t = m & 2047;
          int d = n >> 5, kk = (n >> 4) & 1, h = n & 15;
          if (kk == 0)
            Qout[(((size_t)(b * 16 + h)) * 2048 + t) * 64 + d] = (__bf16)v;
          else  // V stored transposed: [B,H,hd,T]
            Vout[(((size_t)(b * 16 + h)) * 64 + d) * 2048 + t] = (__bf16)v;
        } else {
          Cout[(size_t)m * 1024 + n] = v + bias[n];
        }
      }
}

// ---------------------------------------------------------------------------
// Flash-style attention over one (b,h): 64 query rows per block x 2048 keys.
// score = (2*q.k - |q|^2 - |k|^2) * 0.125 ; keys == queries in this model.
// V pre-transposed in global ([B,H,hd,T]) -> staging is pure b128 async DMA.
// Double-buffered: key-tile kt+64 streams in while tile kt is computed.
__global__ __launch_bounds__(128) void attn_kernel(
    const __bf16* __restrict__ Q, const __bf16* __restrict__ Vt_g,
    const float* __restrict__ sq, __bf16* __restrict__ O) {
  __shared__ __bf16 Ks[2][64 * 64];       // key tile, [key][d]
  __shared__ __bf16 Vt[2][64 * 64];       // value tile, [d][key]
  __shared__ __bf16 Pls[4 * 16 * 64];     // per-wave P staging, [m][s]

  const int bh = blockIdx.y;              // b*16 + h
  const int q0 = blockIdx.x * 64;
  const __bf16* Qh  = Q    + (size_t)bh * 2048 * 64;
  const __bf16* Vth = Vt_g + (size_t)bh * 64 * 2048;
  const float*  sqh = sq   + (size_t)bh * 2048;

  const int tid = threadIdx.x;
  const int wave = tid >> 5, lane = tid & 31;
  const int hi = lane >> 4, lo = lane & 15;

  // Q fragments for this wave's 16 rows (hd=64 -> two K=32 chunks)
  v16bf qf0 = frag_a(Qh, 64, q0 + wave * 16, 0, lane);
  v16bf qf1 = frag_a(Qh, 64, q0 + wave * 16, 32, lane);
  float sqq[8];
  #pragma unroll
  for (int r = 0; r < 8; ++r) sqq[r] = sqh[q0 + wave * 16 + r + 8 * hi];

  v8f o[4];
  float mi[8], li[8];
  #pragma unroll
  for (int j = 0; j < 4; ++j)
    #pragma unroll
    for (int r = 0; r < 8; ++r) o[j][r] = 0.f;
  #pragma unroll
  for (int r = 0; r < 8; ++r) { mi[r] = -1e30f; li[r] = 0.f; }

  auto stage = [&](int kt, int b) {
    #pragma unroll
    for (int i = 0; i < 4; ++i) {          // 64x64 keys + 64x64 V^T
      int idx = tid + i * 128;
      int r = idx >> 3, c = (idx & 7) * 8;
      stage_b128(&Qh[(size_t)(kt + r) * 64 + c], &Ks[b][r * 64 + c]);
      stage_b128(&Vth[(size_t)r * 2048 + kt + c], &Vt[b][r * 64 + c]);
    }
  };

  auto tile_compute = [&](int kt, int bbuf) {
    // S = Q K^T  (16x64 per wave, 8 WMMA)
    v8f s[4];
    #pragma unroll
    for (int j = 0; j < 4; ++j)
      #pragma unroll
      for (int r = 0; r < 8; ++r) s[j][r] = 0.f;
    #pragma unroll
    for (int j = 0; j < 4; ++j) {
      s[j] = wmma_bf16(qf0, frag_b(Ks[bbuf], 64, j * 16, 0, lane), s[j]);
      s[j] = wmma_bf16(qf1, frag_b(Ks[bbuf], 64, j * 16, 32, lane), s[j]);
    }
    float sqk[4];
    #pragma unroll
    for (int j = 0; j < 4; ++j) sqk[j] = sqh[kt + j * 16 + lo];
    #pragma unroll
    for (int j = 0; j < 4; ++j)
      #pragma unroll
      for (int r = 0; r < 8; ++r)
        s[j][r] = (2.f * s[j][r] - sqq[r] - sqk[j]) * 0.125f;

    // online softmax, row reductions across the 16-lane n-group
    #pragma unroll
    for (int r = 0; r < 8; ++r) {
      float mloc = s[0][r];
      #pragma unroll
      for (int j = 1; j < 4; ++j) mloc = fmaxf(mloc, s[j][r]);
      #pragma unroll
      for (int msk = 1; msk < 16; msk <<= 1)
        mloc = fmaxf(mloc, __shfl_xor(mloc, msk, 32));
      float mnew = fmaxf(mi[r], mloc);
      float alpha = __expf(mi[r] - mnew);
      float rsum = 0.f;
      #pragma unroll
      for (int j = 0; j < 4; ++j) {
        float p = __expf(s[j][r] - mnew);
        s[j][r] = p;
        rsum += p;
      }
      #pragma unroll
      for (int msk = 1; msk < 16; msk <<= 1)
        rsum += __shfl_xor(rsum, msk, 32);
      li[r] = li[r] * alpha + rsum;
      mi[r] = mnew;
      #pragma unroll
      for (int j = 0; j < 4; ++j) o[j][r] *= alpha;
    }

    // P (D-layout) -> wave-private LDS row-major, then re-load as A fragments
    __bf16* Pw = Pls + wave * (16 * 64);
    #pragma unroll
    for (int j = 0; j < 4; ++j)
      #pragma unroll
      for (int r = 0; r < 8; ++r)
        Pw[(r + 8 * hi) * 64 + j * 16 + lo] = (__bf16)s[j][r];
    asm volatile("s_wait_dscnt 0" ::: "memory");   // same-wave LDS RAW

    // O += P V  (8 WMMA)
    #pragma unroll
    for (int c = 0; c < 2; ++c) {
      v16bf pa = frag_a(Pw, 64, 0, c * 32, lane);
      #pragma unroll
      for (int j = 0; j < 4; ++j)
        o[j] = wmma_bf16(pa, frag_b(Vt[bbuf], 64, j * 16, c * 32, lane), o[j]);
    }
  };

  stage(0, 0);                               // prologue: fill buffer 0
  for (int kt = 0; kt < 2048; kt += 128) {
    stage_wait();                            // buf0 tile resident
    __syncthreads();                         // all waves done with buf1
    stage(kt + 64, 1);                       // DMA next tile into buf1
    tile_compute(kt, 0);                     // WMMA/softmax on buf0
    stage_wait();
    __syncthreads();
    if (kt + 128 < 2048) stage(kt + 128, 0);
    tile_compute(kt + 64, 1);
  }

  // normalize, write O into [b, t, h, d] (bf16) for the projection GEMM
  const int h = bh & 15, b = bh >> 4;
  #pragma unroll
  for (int r = 0; r < 8; ++r) {
    float inv = 1.f / li[r];
    int m = r + 8 * hi;
    int t = q0 + wave * 16 + m;
    #pragma unroll
    for (int j = 0; j < 4; ++j) {
      int d = j * 16 + lo;
      O[(((size_t)(b * 2048 + t)) * 16 + h) * 64 + d] = (__bf16)(o[j][r] * inv);
    }
  }
}

// ---------------------------------------------------------------------------
extern "C" void kernel_launch(void* const* d_in, const int* in_sizes, int n_in,
                              void* d_out, int out_size, void* d_ws, size_t ws_size,
                              hipStream_t stream) {
  const float* x      = (const float*)d_in[0];   // [2,2048,1024]
  const float* w_qkv  = (const float*)d_in[1];   // [2048,1024]
  const float* w_proj = (const float*)d_in[2];   // [1024,1024]
  const float* b_proj = (const float*)d_in[3];   // [1024]

  char* ws = (char*)d_ws;
  size_t off = 0;
  auto alloc = [&](size_t bytes) -> void* {
    void* p = ws + off;
    off += (bytes + 255) & ~(size_t)255;
    return p;
  };
  __bf16* xbf     = (__bf16*)alloc(4096ull * 1024 * 2);
  __bf16* wqkvbf  = (__bf16*)alloc(2048ull * 1024 * 2);
  __bf16* wprojbf = (__bf16*)alloc(1024ull * 1024 * 2);
  __bf16* Qbf     = (__bf16*)alloc(32ull * 2048 * 64 * 2);
  __bf16* Vtbf    = (__bf16*)alloc(32ull * 64 * 2048 * 2);   // [B,H,hd,T]
  float*  sq      = (float*)alloc(32ull * 2048 * 4);
  __bf16* Obf     = (__bf16*)alloc(4096ull * 1024 * 2);

  convert_bf16<<<(4194304 + 255) / 256, 256, 0, stream>>>(x, xbf, 4194304);
  convert_bf16<<<(2097152 + 255) / 256, 256, 0, stream>>>(w_qkv, wqkvbf, 2097152);
  convert_bf16<<<(1048576 + 255) / 256, 256, 0, stream>>>(w_proj, wprojbf, 1048576);

  // qkv = x @ w_qkv^T, scattered into Q [B,H,T,hd] and V^T [B,H,hd,T]
  gemm128x64<0><<<dim3(2048 / 64, 4096 / 128), 128, 0, stream>>>(
      xbf, wqkvbf, Qbf, Vtbf, nullptr, nullptr);

  sq_kernel<<<65536 / 256, 256, 0, stream>>>(Qbf, sq);

  attn_kernel<<<dim3(2048 / 64, 32), 128, 0, stream>>>(Qbf, Vtbf, sq, Obf);

  // out = O @ w_proj^T + b_proj  (f32)
  gemm128x64<1><<<dim3(1024 / 64, 4096 / 128), 128, 0, stream>>>(
      Obf, wprojbf, nullptr, nullptr, b_proj, (float*)d_out);
}
